// GNNDispatchPolicy_38199439130925
// MI455X (gfx1250) — matrix operand order repaired and checked by reading, not compiled, not run on hardware
//
#include <hip/hip_runtime.h>
#include <hip/hip_bf16.h>

#define NN 30000        // nodes
#define NE 500000       // edges
#define NP 2000         // patients
#define ND 50           // depots
#define HD 128          // hidden
#define LN_EPS 1e-5f

typedef _Float16 half_t;
typedef __attribute__((ext_vector_type(16))) _Float16 v16h;
typedef __attribute__((ext_vector_type(8)))  float    v8f;

// ---------------------------------------------------------------------------
// WMMA helpers (gfx1250, wave32): D = A(16x32 f16) * B(32x16 f16) + C(16x16 f32)
// ---------------------------------------------------------------------------
__device__ __forceinline__ v8f wmma_f16(v16h a, v16h b, v8f c) {
  return __builtin_amdgcn_wmma_f32_16x16x32_f16(false, a, false, b, (short)0, c,
                                                false, false);
}

// A fragment: p already points at base + row*lda + kb; elems {0..15} contiguous
__device__ __forceinline__ v16h load_a_frag(const half_t* p) {
  v16h a;
#pragma unroll
  for (int i = 0; i < 8; ++i) a[i] = p[i];
#pragma unroll
  for (int i = 0; i < 8; ++i) a[8 + i] = p[8 + i];
  return a;
}

// B fragment: 16 contiguous f16 (Wt is N-major with K=128 rows per col)
__device__ __forceinline__ v16h load_b16(const half_t* p) {
  v16h b;
#pragma unroll
  for (int i = 0; i < 16; ++i) b[i] = p[i];
  return b;
}

// ---------------------------------------------------------------------------
// Small utility kernels
// ---------------------------------------------------------------------------
__global__ void k_zero_f32(float* p, int n) {
  int i = blockIdx.x * 256 + threadIdx.x;
  if (i < n) p[i] = 0.f;
}

// Wt[n*128 + k] = (f16) W[k*128 + n]   (128x128 only)
__global__ void k_transpose16(const float* __restrict__ W, half_t* __restrict__ Wt) {
  int i = blockIdx.x * 256 + threadIdx.x;
  if (i >= HD * HD) return;
  int n = i >> 7, k = i & 127;
  Wt[i] = (half_t)W[k * HD + n];
}

__global__ void k_deg(const int* __restrict__ dst, float* deg, int E) {
  int i = blockIdx.x * 256 + threadIdx.x;
  if (i < E) atomicAdd(deg + dst[i], 1.f);
}

__global__ void k_aggnorm(const float* __restrict__ agg, const float* __restrict__ deg,
                          half_t* __restrict__ out, int n) {
  int i = blockIdx.x * 256 + threadIdx.x;
  if (i >= n) return;
  float d = fmaxf(deg[i >> 7], 1.f);
  out[i] = (half_t)(agg[i] / d);
}

// encoder layer 1: out = relu(nf @ W1 + b1), nf is M x 16, out f16 M x 128
__global__ void k_enc1(const float* __restrict__ nf, const float* __restrict__ W1,
                       const float* __restrict__ b1, half_t* __restrict__ out, int M) {
  int i = blockIdx.x * 256 + threadIdx.x;
  if (i >= M * HD) return;
  int n = i >> 7, c = i & 127;
  const float* x = nf + n * 16;
  float s = b1[c];
#pragma unroll
  for (int j = 0; j < 16; ++j) s += x[j] * W1[j * HD + c];
  out[i] = (half_t)fmaxf(s, 0.f);
}

// ---------------------------------------------------------------------------
// Generic WMMA GEMM (K = 128, N = 128): out = [relu](A1@W1 [+ A2@W2] + bias)
// Block: 128 thr = 4 waves, tile 64 rows x 128 cols; wave = 32 rows x 64 cols.
// Each B fragment feeds 2 WMMAs (2 A row-fragments).
// ---------------------------------------------------------------------------
__global__ __launch_bounds__(128) void k_gemm128(
    const half_t* __restrict__ A1, const half_t* __restrict__ W1t,
    const half_t* __restrict__ A2, const half_t* __restrict__ W2t,
    const int* __restrict__ rowidx, const float* __restrict__ bias, int relu,
    float* __restrict__ outF, half_t* __restrict__ outH, int M) {
  __shared__ half_t As[64 * 32];
  const int m0 = blockIdx.x * 64;
  const int tid = threadIdx.x;
  const int wave = tid >> 5, lane = tid & 31;
  const int rw = wave & 1, cg = wave >> 1;            // rows rw*32.., cols cg*64..
  const int aoff = (lane & 15) * 32 + ((lane < 16) ? 0 : 16);
  const int kbb = (lane < 16) ? 0 : 16;
  const int bcol = lane & 15;
  const bool full = (m0 + 64 <= M) && (rowidx == nullptr);
  v8f acc[2][4] = {};

  for (int part = 0; part < 2; ++part) {
    const half_t* A = part ? A2 : A1;
    const half_t* Wt = part ? W2t : W1t;
    if (!A) continue;
    for (int k0 = 0; k0 < 128; k0 += 32) {
      __syncthreads();
      if (full) {
        const half_t* Ab = A + m0 * HD + k0;
#pragma unroll
        for (int i = 0; i < 4; ++i) {
          int idx = tid + i * 128;                    // 0..511 uint2 slots
          int r = idx >> 3, c = (idx & 7) * 4;
          *(uint2*)(As + r * 32 + c) = *(const uint2*)(Ab + r * HD + c);
        }
      } else {
#pragma unroll
        for (int i = 0; i < 4; ++i) {
          int idx = tid + i * 128;
          int r = idx >> 3, c = (idx & 7) * 4;
          int gr = m0 + r;
          uint2 v; v.x = 0u; v.y = 0u;
          if (gr < M) {
            int row = rowidx ? rowidx[gr] : gr;
            v = *(const uint2*)(A + row * HD + k0 + c);
          }
          *(uint2*)(As + r * 32 + c) = v;
        }
      }
      __syncthreads();
      v16h a0 = load_a_frag(As + (rw * 32) * 32 + aoff);
      v16h a1 = load_a_frag(As + (rw * 32 + 16) * 32 + aoff);
#pragma unroll
      for (int ct = 0; ct < 4; ++ct) {
        v16h b = load_b16(Wt + ((cg * 64 + ct * 16 + bcol) << 7) + k0 + kbb);
        acc[0][ct] = wmma_f16(a0, b, acc[0][ct]);
        acc[1][ct] = wmma_f16(a1, b, acc[1][ct]);
      }
    }
  }

  const int rofs = (lane >= 16) ? 8 : 0, c0 = lane & 15;
#pragma unroll
  for (int rr = 0; rr < 2; ++rr)
#pragma unroll
    for (int ct = 0; ct < 4; ++ct) {
      int col = cg * 64 + ct * 16 + c0;
#pragma unroll
      for (int v = 0; v < 8; ++v) {
        int r = m0 + rw * 32 + rr * 16 + v + rofs;
        if (r < M) {
          float x = acc[rr][ct][v] + bias[col];
          if (relu) x = fmaxf(x, 0.f);
          if (outF) outF[r * HD + col] = x;
          if (outH) outH[r * HD + col] = (half_t)x;
        }
      }
    }
}

// ---------------------------------------------------------------------------
// Fused per-edge message MLP + scatter-add. 64 edges / 128-thread block.
//   hidden = relu(h[src]@Ws + h[dst]@Wd + ef@We + b1)
//   agg[dst] += hidden @ W2 + b2   (f32 atomics)
// ---------------------------------------------------------------------------
__global__ __launch_bounds__(128) void k_edge_msg(
    const half_t* __restrict__ h16, const int* __restrict__ src,
    const int* __restrict__ dst, const float* __restrict__ ef,
    const half_t* __restrict__ Wst, const half_t* __restrict__ Wdt,
    const float* __restrict__ We, const float* __restrict__ b1,
    const half_t* __restrict__ W2t, const float* __restrict__ b2,
    float* __restrict__ agg, int E) {
  __shared__ half_t Hs[64 * HD];
  __shared__ half_t Hd[64 * HD];
  __shared__ half_t Hid[64 * HD];
  __shared__ float EF[64 * 4];
  const int e0 = blockIdx.x * 64;
  const int tid = threadIdx.x;

  for (int i = tid; i < 1024; i += 128) {          // 64 rows x 16 uint4
    int r = i >> 4, c4 = i & 15;
    int e = e0 + r;
    int s = 0, d = 0;
    if (e < E) { s = src[e]; d = dst[e]; }
    ((uint4*)Hs)[i] = ((const uint4*)(h16 + s * HD))[c4];
    ((uint4*)Hd)[i] = ((const uint4*)(h16 + d * HD))[c4];
  }
  for (int i = tid; i < 256; i += 128) {
    int e = e0 + (i >> 2);
    EF[i] = (e < E) ? ef[e * 4 + (i & 3)] : 0.f;
  }
  __syncthreads();

  const int wave = tid >> 5, lane = tid & 31;
  const int rw = wave & 1, cg = wave >> 1;
  const int aoff = (lane & 15) * HD + ((lane < 16) ? 0 : 16);
  const int kbb = (lane < 16) ? 0 : 16;
  const int bcol = lane & 15;
  const int rofs = (lane >= 16) ? 8 : 0, c0 = lane & 15;
  const half_t* Hsb = Hs + rw * 32 * HD;
  const half_t* Hdb = Hd + rw * 32 * HD;

  v8f acc[2][4] = {};
#pragma unroll
  for (int k0 = 0; k0 < 128; k0 += 32) {
    v16h a0 = load_a_frag(Hsb + aoff + k0);
    v16h a1 = load_a_frag(Hsb + 16 * HD + aoff + k0);
#pragma unroll
    for (int ct = 0; ct < 4; ++ct) {
      v16h b = load_b16(Wst + ((cg * 64 + ct * 16 + bcol) << 7) + k0 + kbb);
      acc[0][ct] = wmma_f16(a0, b, acc[0][ct]);
      acc[1][ct] = wmma_f16(a1, b, acc[1][ct]);
    }
    v16h a2 = load_a_frag(Hdb + aoff + k0);
    v16h a3 = load_a_frag(Hdb + 16 * HD + aoff + k0);
#pragma unroll
    for (int ct = 0; ct < 4; ++ct) {
      v16h b = load_b16(Wdt + ((cg * 64 + ct * 16 + bcol) << 7) + k0 + kbb);
      acc[0][ct] = wmma_f16(a2, b, acc[0][ct]);
      acc[1][ct] = wmma_f16(a3, b, acc[1][ct]);
    }
  }
#pragma unroll
  for (int rr = 0; rr < 2; ++rr)
#pragma unroll
    for (int ct = 0; ct < 4; ++ct) {
      int col = cg * 64 + ct * 16 + c0;
#pragma unroll
      for (int v = 0; v < 8; ++v) {
        int r = rw * 32 + rr * 16 + v + rofs;
        const float* e4 = EF + r * 4;
        float x = acc[rr][ct][v] + b1[col] + e4[0] * We[col] + e4[1] * We[HD + col] +
                  e4[2] * We[2 * HD + col] + e4[3] * We[3 * HD + col];
        Hid[r * HD + col] = (half_t)fmaxf(x, 0.f);
      }
    }
  __syncthreads();

  v8f acc2[2][4] = {};
  const half_t* Hb = Hid + rw * 32 * HD;
#pragma unroll
  for (int k0 = 0; k0 < 128; k0 += 32) {
    v16h a0 = load_a_frag(Hb + aoff + k0);
    v16h a1 = load_a_frag(Hb + 16 * HD + aoff + k0);
#pragma unroll
    for (int ct = 0; ct < 4; ++ct) {
      v16h b = load_b16(W2t + ((cg * 64 + ct * 16 + bcol) << 7) + k0 + kbb);
      acc2[0][ct] = wmma_f16(a0, b, acc2[0][ct]);
      acc2[1][ct] = wmma_f16(a1, b, acc2[1][ct]);
    }
  }
#pragma unroll
  for (int rr = 0; rr < 2; ++rr)
#pragma unroll
    for (int ct = 0; ct < 4; ++ct) {
      int col = cg * 64 + ct * 16 + c0;
#pragma unroll
      for (int v = 0; v < 8; ++v) {
        int r = rw * 32 + rr * 16 + v + rofs;
        int e = e0 + r;
        if (e < E) atomicAdd(agg + dst[e] * HD + col, acc2[rr][ct][v] + b2[col]);
      }
    }
}

// ---------------------------------------------------------------------------
// LayerNorm over 128 features; one node per 128-thread block.
// ---------------------------------------------------------------------------
__global__ void k_layernorm(const float* __restrict__ X, const float* __restrict__ g,
                            const float* __restrict__ b, float* __restrict__ outF,
                            half_t* __restrict__ outH) {
  __shared__ float s[128];
  int n = blockIdx.x, t = threadIdx.x;
  float x = X[n * HD + t];
  s[t] = x;
  __syncthreads();
  for (int o = 64; o > 0; o >>= 1) { if (t < o) s[t] += s[t + o]; __syncthreads(); }
  float mu = s[0] * (1.f / HD);
  __syncthreads();
  float d = x - mu;
  s[t] = d * d;
  __syncthreads();
  for (int o = 64; o > 0; o >>= 1) { if (t < o) s[t] += s[t + o]; __syncthreads(); }
  float var = s[0] * (1.f / HD);
  float y = d * rsqrtf(var + LN_EPS) * g[t] + b[t];
  if (outF) outF[n * HD + t] = y;
  outH[n * HD + t] = (half_t)y;
}

// out[r] = dot(Hin[r,:], w2) + b2   (4 threads / row, 64 rows / block)
__global__ void k_dot_head(const half_t* __restrict__ Hin, const float* __restrict__ w2,
                           const float* __restrict__ b2, float* __restrict__ out,
                           int M) {
  __shared__ float red[256];
  int tid = threadIdx.x;
  int r = blockIdx.x * 64 + (tid >> 2), q = tid & 3;
  float part = 0.f;
  if (r < M) {
    const half_t* hr = Hin + r * HD + q * 32;
#pragma unroll
    for (int j = 0; j < 32; ++j) part += (float)hr[j] * w2[q * 32 + j];
  }
  red[tid] = part;
  __syncthreads();
  if ((tid & 3) == 0 && r < M)
    out[r] = red[tid] + red[tid + 1] + red[tid + 2] + red[tid + 3] + b2[0];
}

// ---------------------------------------------------------------------------
// Assignment head: hidden = relu(h[p]@Ws + h[d]@Wd + pf@We + b1); s = hidden.w2+b2
// Writes scores[i] and logits[p*ND + (d-NP)].  64 pairs / 128-thread block.
// ---------------------------------------------------------------------------
__global__ __launch_bounds__(128) void k_pair(
    const half_t* __restrict__ h16, const int* __restrict__ pair,
    const float* __restrict__ pf, const half_t* __restrict__ Wst,
    const half_t* __restrict__ Wdt, const float* __restrict__ We,
    const float* __restrict__ b1, const float* __restrict__ w2,
    const float* __restrict__ b2, float* __restrict__ logits,
    float* __restrict__ scores, int M) {
  __shared__ half_t Hs[64 * HD];
  __shared__ half_t Hd[64 * HD];
  __shared__ half_t Hid[64 * HD];
  __shared__ float PF[64 * 4];
  __shared__ float red[128];
  const int e0 = blockIdx.x * 64;
  const int tid = threadIdx.x;

  for (int i = tid; i < 1024; i += 128) {
    int r = i >> 4, c4 = i & 15;
    int e = e0 + r;
    int p = 0, d = 0;
    if (e < M) { p = pair[2 * e]; d = pair[2 * e + 1]; }
    ((uint4*)Hs)[i] = ((const uint4*)(h16 + p * HD))[c4];
    ((uint4*)Hd)[i] = ((const uint4*)(h16 + d * HD))[c4];
  }
  for (int i = tid; i < 256; i += 128) {
    int e = e0 + (i >> 2);
    PF[i] = (e < M) ? pf[e * 4 + (i & 3)] : 0.f;
  }
  __syncthreads();

  const int wave = tid >> 5, lane = tid & 31;
  const int rw = wave & 1, cg = wave >> 1;
  const int aoff = (lane & 15) * HD + ((lane < 16) ? 0 : 16);
  const int kbb = (lane < 16) ? 0 : 16;
  const int bcol = lane & 15;
  const int rofs = (lane >= 16) ? 8 : 0, c0 = lane & 15;
  const half_t* Hsb = Hs + rw * 32 * HD;
  const half_t* Hdb = Hd + rw * 32 * HD;

  v8f acc[2][4] = {};
#pragma unroll
  for (int k0 = 0; k0 < 128; k0 += 32) {
    v16h a0 = load_a_frag(Hsb + aoff + k0);
    v16h a1 = load_a_frag(Hsb + 16 * HD + aoff + k0);
#pragma unroll
    for (int ct = 0; ct < 4; ++ct) {
      v16h b = load_b16(Wst + ((cg * 64 + ct * 16 + bcol) << 7) + k0 + kbb);
      acc[0][ct] = wmma_f16(a0, b, acc[0][ct]);
      acc[1][ct] = wmma_f16(a1, b, acc[1][ct]);
    }
    v16h a2 = load_a_frag(Hdb + aoff + k0);
    v16h a3 = load_a_frag(Hdb + 16 * HD + aoff + k0);
#pragma unroll
    for (int ct = 0; ct < 4; ++ct) {
      v16h b = load_b16(Wdt + ((cg * 64 + ct * 16 + bcol) << 7) + k0 + kbb);
      acc[0][ct] = wmma_f16(a2, b, acc[0][ct]);
      acc[1][ct] = wmma_f16(a3, b, acc[1][ct]);
    }
  }
#pragma unroll
  for (int rr = 0; rr < 2; ++rr)
#pragma unroll
    for (int ct = 0; ct < 4; ++ct) {
      int col = cg * 64 + ct * 16 + c0;
#pragma unroll
      for (int v = 0; v < 8; ++v) {
        int r = rw * 32 + rr * 16 + v + rofs;
        const float* e4 = PF + r * 4;
        float x = acc[rr][ct][v] + b1[col] + e4[0] * We[col] + e4[1] * We[HD + col] +
                  e4[2] * We[2 * HD + col] + e4[3] * We[3 * HD + col];
        Hid[r * HD + col] = (half_t)fmaxf(x, 0.f);
      }
    }
  __syncthreads();

  {
    int r = tid >> 1, q = tid & 1;
    const half_t* hr = Hid + r * HD + q * 64;
    float part = 0.f;
#pragma unroll
    for (int j = 0; j < 64; ++j) part += (float)hr[j] * w2[q * 64 + j];
    red[tid] = part;
  }
  __syncthreads();
  if ((tid & 1) == 0) {
    int r = tid >> 1;
    int e = e0 + r;
    if (e < M) {
      float s = red[tid] + red[tid + 1] + b2[0];
      scores[e] = s;
      int p = pair[2 * e], dl = pair[2 * e + 1] - NP;
      if (p >= 0 && p < NP && dl >= 0 && dl < ND) logits[p * ND + dl] = s;
    }
  }
}

// ---------------------------------------------------------------------------
// Host launcher
// ---------------------------------------------------------------------------
extern "C" void kernel_launch(void* const* d_in, const int* in_sizes, int n_in,
                              void* d_out, int out_size, void* d_ws, size_t ws_size,
                              hipStream_t stream) {
  (void)in_sizes; (void)n_in; (void)out_size; (void)ws_size;
  const float* nf   = (const float*)d_in[0];
  const int*   eidx = (const int*)d_in[1];     // (2, NE)
  const float* ef   = (const float*)d_in[2];
  const int*   pidx = (const int*)d_in[3];     // patient indices (2000)
  const int*   pair = (const int*)d_in[5];     // (NP*ND, 2)
  const float* pf   = (const float*)d_in[6];
  const int* src = eidx;
  const int* dst = eidx + NE;

  // params in jax pytree (sorted-key) order, starting at index 7
  const float* asg_W1 = (const float*)d_in[7];
  const float* asg_W2 = (const float*)d_in[8];
  const float* asg_b1 = (const float*)d_in[9];
  const float* asg_b2 = (const float*)d_in[10];
  struct ConvP { const float *ln_b, *ln_g, *msg_W1, *msg_W2, *msg_b1, *msg_b2,
                              *upd_W1, *upd_W2, *upd_b1, *upd_b2; } cv[3];
  for (int l = 0; l < 3; ++l) {
    int b = 11 + 10 * l;
    cv[l].ln_b   = (const float*)d_in[b + 0];
    cv[l].ln_g   = (const float*)d_in[b + 1];
    cv[l].msg_W1 = (const float*)d_in[b + 2];
    cv[l].msg_W2 = (const float*)d_in[b + 3];
    cv[l].msg_b1 = (const float*)d_in[b + 4];
    cv[l].msg_b2 = (const float*)d_in[b + 5];
    cv[l].upd_W1 = (const float*)d_in[b + 6];
    cv[l].upd_W2 = (const float*)d_in[b + 7];
    cv[l].upd_b1 = (const float*)d_in[b + 8];
    cv[l].upd_b2 = (const float*)d_in[b + 9];
  }
  const float* enc_W1 = (const float*)d_in[41];
  const float* enc_W2 = (const float*)d_in[42];
  const float* enc_b1 = (const float*)d_in[43];
  const float* enc_b2 = (const float*)d_in[44];
  const float* pri_W1 = (const float*)d_in[45];
  const float* pri_W2 = (const float*)d_in[46];
  const float* pri_b1 = (const float*)d_in[47];
  const float* pri_b2 = (const float*)d_in[48];

  // workspace bump allocator
  char* ws = (char*)d_ws;
  size_t off = 0;
  auto alloc = [&](size_t bytes) -> void* {
    void* p = ws + off;
    off += (bytes + 255) & ~(size_t)255;
    return p;
  };
  half_t* h16a   = (half_t*)alloc((size_t)NN * HD * 2);
  half_t* h16b   = (half_t*)alloc((size_t)NN * HD * 2);
  half_t* t16    = (half_t*)alloc((size_t)NN * HD * 2);
  half_t* aggn16 = (half_t*)alloc((size_t)NN * HD * 2);
  float*  agg    = (float*)alloc((size_t)NN * HD * 4);
  float*  u32b   = (float*)alloc((size_t)NN * HD * 4);
  float*  deg    = (float*)alloc((size_t)NN * 4);
  half_t* prih16 = (half_t*)alloc((size_t)NP * HD * 2);
  const size_t WB = (size_t)HD * HD * 2;   // bytes per 128x128 f16 transpose
  half_t* encW2t = (half_t*)alloc(WB);
  half_t* priW1t = (half_t*)alloc(WB);
  half_t* asgWst = (half_t*)alloc(WB);
  half_t* asgWdt = (half_t*)alloc(WB);
  half_t *cWs[3], *cWd[3], *cW2[3], *cWh[3], *cWa[3], *cU2[3];
  for (int l = 0; l < 3; ++l) {
    cWs[l] = (half_t*)alloc(WB);
    cWd[l] = (half_t*)alloc(WB);
    cW2[l] = (half_t*)alloc(WB);
    cWh[l] = (half_t*)alloc(WB);
    cWa[l] = (half_t*)alloc(WB);
    cU2[l] = (half_t*)alloc(WB);
  }

  dim3 b256(256), b128(128);
  const dim3 gT((HD * HD + 255) / 256);
  auto TR = [&](const float* W, half_t* Wt) {
    k_transpose16<<<gT, b256, 0, stream>>>(W, Wt);
  };
  TR(enc_W2, encW2t);
  TR(pri_W1, priW1t);
  TR(asg_W1, asgWst);                    // rows 0..127 of (260x128)
  TR(asg_W1 + 128 * HD, asgWdt);         // rows 128..255
  for (int l = 0; l < 3; ++l) {
    TR(cv[l].msg_W1, cWs[l]);
    TR(cv[l].msg_W1 + 128 * HD, cWd[l]);
    TR(cv[l].msg_W2, cW2[l]);
    TR(cv[l].upd_W1, cWh[l]);
    TR(cv[l].upd_W1 + 128 * HD, cWa[l]);
    TR(cv[l].upd_W2, cU2[l]);
  }

  float* out_h      = (float*)d_out;                    // NN*HD
  float* out_pri    = out_h + (size_t)NN * HD;          // NP
  float* out_logits = out_pri + NP;                     // NP*ND
  float* out_scores = out_logits + (size_t)NP * ND;     // NP*ND

  // encoder
  k_enc1<<<dim3((NN * HD + 255) / 256), b256, 0, stream>>>(nf, enc_W1, enc_b1, t16, NN);
  k_gemm128<<<dim3((NN + 63) / 64), b128, 0, stream>>>(
      t16, encW2t, nullptr, nullptr, nullptr, enc_b2, 0, nullptr, h16a, NN);

  // degrees (layer-invariant)
  k_zero_f32<<<dim3((NN + 255) / 256), b256, 0, stream>>>(deg, NN);
  k_deg<<<dim3((NE + 255) / 256), b256, 0, stream>>>(dst, deg, NE);

  half_t* hcur = h16a;
  half_t* hnxt = h16b;
  for (int l = 0; l < 3; ++l) {
    k_zero_f32<<<dim3((NN * HD + 255) / 256), b256, 0, stream>>>(agg, NN * HD);
    k_edge_msg<<<dim3((NE + 63) / 64), b128, 0, stream>>>(
        hcur, src, dst, ef, cWs[l], cWd[l], cv[l].msg_W1 + 256 * HD, cv[l].msg_b1,
        cW2[l], cv[l].msg_b2, agg, NE);
    k_aggnorm<<<dim3((NN * HD + 255) / 256), b256, 0, stream>>>(agg, deg, aggn16,
                                                               NN * HD);
    // update MLP: hidden = relu(h@Wh + agg@Wa + b1) ; u = hidden@W2 + b2
    k_gemm128<<<dim3((NN + 63) / 64), b128, 0, stream>>>(
        hcur, cWh[l], aggn16, cWa[l], nullptr, cv[l].upd_b1, 1, nullptr, t16, NN);
    k_gemm128<<<dim3((NN + 63) / 64), b128, 0, stream>>>(
        t16, cU2[l], nullptr, nullptr, nullptr, cv[l].upd_b2, 0, u32b, nullptr, NN);
    k_layernorm<<<dim3(NN), b128, 0, stream>>>(
        u32b, cv[l].ln_g, cv[l].ln_b, (l == 2) ? out_h : nullptr, hnxt);
    half_t* tmp = hcur; hcur = hnxt; hnxt = tmp;
  }

  // priority head
  k_gemm128<<<dim3((NP + 63) / 64), b128, 0, stream>>>(
      hcur, priW1t, nullptr, nullptr, pidx, pri_b1, 1, nullptr, prih16, NP);
  k_dot_head<<<dim3((NP + 63) / 64), b256, 0, stream>>>(prih16, pri_W2, pri_b2,
                                                        out_pri, NP);

  // assignment head
  k_pair<<<dim3((NP * ND + 63) / 64), b128, 0, stream>>>(
      hcur, pair, pf, asgWst, asgWdt, asg_W1 + 256 * HD, asg_b1, asg_W2, asg_b2,
      out_logits, out_scores, NP * ND);
}
